// PointSIFT_4389456577478
// MI455X (gfx1250) — compile-verified, matrix operand order checked by reading, not run on hardware
//
#include <hip/hip_runtime.h>
#include <math.h>

// ---------------- problem constants ----------------
constexpr int BZ = 4;          // batch
constexpr int NN = 4096;       // points
constexpr int LOG2N = 12;
constexpr int OC = 64;         // channels
constexpr float JUDGE = 0.1f * 0.1f;

constexpr long M1 = (long)BZ * NN * 4;   // 65536
constexpr long M2 = (long)BZ * NN * 2;   // 32768
constexpr long M3 = (long)BZ * NN;       // 16384

// ---------------- workspace layout (bytes) ----------------
constexpr size_t OFF_IDX = 0;                         // B*N*8 ints        = 524288
constexpr size_t OFF_W   = 524288;                    // 6 * 64*160 f16    = 122880
constexpr size_t OFF_ST  = 647168;                    // 256 floats        = 1024
constexpr size_t OFF_XA  = (size_t)1 << 20;           // 65536*160 f16     = 20971520
constexpr size_t OFF_XB  = OFF_XA + 20971520;         // 32768*128 f16     = 8388608
constexpr size_t OFF_XC  = OFF_XB + 8388608;          // 16384*128 f16     = 4194304
constexpr size_t OFF_NP  = OFF_XC + 4194304;          // 16384*64 f16      = 2097152
constexpr size_t OFF_Y   = OFF_NP + 2097152;          // 65536*64 f32      = 16777216
constexpr size_t WS_NEED = OFF_Y + 16777216;          // ~53.5 MB

typedef __attribute__((ext_vector_type(16))) _Float16 v16h;
typedef __attribute__((ext_vector_type(8)))  _Float16 v8h;
typedef __attribute__((ext_vector_type(8)))  float    v8f;

// =====================================================================
// 1) Neighbor select: per-centroid scan over all candidates (LDS-tiled),
//    branchless 8-octant running min kept in registers.
// =====================================================================
__global__ __launch_bounds__(256) void select_kernel(const float* __restrict__ xyz,
                                                     int* __restrict__ idx) {
  __shared__ float sx[256], sy[256], sz[256];
  const int b = blockIdx.y;
  const int n = blockIdx.x * 256 + threadIdx.x;
  const float* xb = xyz + (size_t)b * NN * 3;
  const float cx = xb[n * 3 + 0], cy = xb[n * 3 + 1], cz = xb[n * 3 + 2];

  float best[8];
  int   bi[8];
#pragma unroll
  for (int o = 0; o < 8; ++o) { best[o] = JUDGE; bi[o] = n; }  // diagonal seeds judge

  for (int t0 = 0; t0 < NN; t0 += 256) {
    const int m0 = t0 + threadIdx.x;
    sx[threadIdx.x] = xb[m0 * 3 + 0];
    sy[threadIdx.x] = xb[m0 * 3 + 1];
    sz[threadIdx.x] = xb[m0 * 3 + 2];
    __syncthreads();
    for (int j = 0; j < 256; ++j) {
      const int m = t0 + j;
      const float dx = sx[j] - cx, dy = sy[j] - cy, dz = sz[j] - cz;
      const float d2 = dx * dx + dy * dy + dz * dz;
      const bool ok = (m != n) & (d2 > 1e-10f) & (d2 < JUDGE);
      // diff in (-1,1): trunc(diff+1) in {0,1}
      const int oct = 4 * (int)(dx + 1.0f) + 2 * (int)(dy + 1.0f) + (int)(dz + 1.0f);
#pragma unroll
      for (int o = 0; o < 8; ++o) {
        const bool u = ok & (oct == o) & (d2 < best[o]);
        best[o] = u ? d2 : best[o];
        bi[o]   = u ? m  : bi[o];
      }
    }
    __syncthreads();
  }
  int* op = idx + ((size_t)b * NN + n) * 8;
#pragma unroll
  for (int o = 0; o < 8; ++o) op[o] = bi[o];
}

// =====================================================================
// 2) Weight prep: W[oc][ic][0][kk] (f32) -> Wt^T[oc][k] f16, k = kk*cin+ic,
//    zero-padded to K=160. (conv bias cancels under BN mean-subtraction.)
// =====================================================================
__global__ void prep_weights(const float* __restrict__ W, _Float16* __restrict__ wt, int cin) {
  const int g = blockIdx.x * 256 + threadIdx.x;
  if (g >= 64 * 160) return;
  const int oc = g / 160, k = g % 160;
  float v = 0.f;
  if (k < 2 * cin) {
    const int kk = k / cin, ic = k % cin;
    v = W[((size_t)oc * cin + ic) * 2 + kk];
  }
  wt[(size_t)oc * 160 + k] = (_Float16)v;
}

// =====================================================================
// 3) im2col for gathered features: row (b,n,w), k = kk*67 + c,
//    feat = [xyz[id]-xyz[n] (3), feats[id] (64)], padded to 160. f16 out.
// =====================================================================
__global__ void im2col_first(const float* __restrict__ xyz, const float* __restrict__ pts,
                             const int* __restrict__ idx, _Float16* __restrict__ X) {
  const long g = (long)blockIdx.x * 256 + threadIdx.x;
  if (g >= M1 * 160) return;
  const int k = (int)(g % 160);
  const long r = g / 160;
  const int w = (int)(r & 3);
  const long bn = r >> 2;
  const int n = (int)(bn & (NN - 1));
  const int b = (int)(bn >> LOG2N);
  float v = 0.f;
  if (k < 134) {
    const int kk = k / 67, c = k % 67;
    const int id = idx[bn * 8 + 2 * w + kk];
    const float* xb = xyz + (size_t)b * NN * 3;
    if (c < 3) v = xb[id * 3 + c] - xb[n * 3 + c];
    else       v = pts[((size_t)b * NN + id) * 64 + (c - 3)];
  }
  X[g] = (_Float16)v;
}

__global__ void im2col_second(const float* __restrict__ xyz, const _Float16* __restrict__ np,
                              const int* __restrict__ idx, _Float16* __restrict__ X) {
  const long g = (long)blockIdx.x * 256 + threadIdx.x;
  if (g >= M1 * 160) return;
  const int k = (int)(g % 160);
  const long r = g / 160;
  const int w = (int)(r & 3);
  const long bn = r >> 2;
  const int n = (int)(bn & (NN - 1));
  const int b = (int)(bn >> LOG2N);
  float v = 0.f;
  if (k < 134) {
    const int kk = k / 67, c = k % 67;
    const int id = idx[bn * 8 + 2 * w + kk];
    const float* xb = xyz + (size_t)b * NN * 3;
    if (c < 3) v = xb[id * 3 + c] - xb[n * 3 + c];
    else       v = (float)np[((size_t)b * NN + id) * 64 + (c - 3)];
  }
  X[g] = (_Float16)v;
}

// =====================================================================
// 4) WMMA GEMM + BN partial stats.
//    Block = 256 thr = 8 waves; wave w -> rows [blk*128 + w*16, +16), all 64 cols
//    as 4 v_wmma_f32_16x16x32_f16 accumulators. Weights^T staged in LDS.
// =====================================================================
__device__ __forceinline__ v16h make_frag(const _Float16* p0, const _Float16* p1) {
  const v8h lo = *(const v8h*)p0;
  const v8h hi = *(const v8h*)p1;
  v16h f;
#pragma unroll
  for (int i = 0; i < 8; ++i) { f[i] = lo[i]; f[i + 8] = hi[i]; }
  return f;
}

__global__ __launch_bounds__(256) void gemm_bn(const _Float16* __restrict__ X,
                                               const _Float16* __restrict__ Wt,
                                               float* __restrict__ Y,
                                               float* __restrict__ gsum, float* __restrict__ gssq,
                                               int ldx, int kIters) {
  __shared__ __align__(16) _Float16 wl[64 * 160];
  __shared__ float s_sum[64], s_ssq[64];
  const int tid = threadIdx.x;
  {  // cooperative weight stage: 20480 B = 1280 uint4
    const uint4* s = (const uint4*)Wt;
    uint4* d = (uint4*)wl;
#pragma unroll
    for (int i = 0; i < 5; ++i) d[tid + 256 * i] = s[tid + 256 * i];
  }
  if (tid < 64) { s_sum[tid] = 0.f; s_ssq[tid] = 0.f; }
  __syncthreads();

  const int wave = tid >> 5, lane = tid & 31;
  const int grp = lane >> 4, lh = lane & 15;
  const long row = (long)blockIdx.x * 128 + wave * 16 + lh;  // A: lane L%16 holds row M
  const _Float16* xr = X + row * (long)ldx;

  v8f acc[4] = {};
  for (int kt = 0; kt < kIters; kt += 32) {
    __builtin_prefetch(xr + kt + 64, 0, 0);
    const v16h a = make_frag(xr + kt + 8 * grp, xr + kt + 16 + 8 * grp);
#pragma unroll
    for (int t = 0; t < 4; ++t) {
      const _Float16* wr = wl + (size_t)(t * 16 + lh) * 160 + kt + 8 * grp;
      const v16h bf = make_frag(wr, wr + 16);
      acc[t] = __builtin_amdgcn_wmma_f32_16x16x32_f16(
          false, a, false, bf, (short)0, acc[t], false, false);
    }
  }

  // D: lane holds col lh, rows 8*grp + r within the wave's 16-row tile
  const long mb = (long)blockIdx.x * 128 + wave * 16 + 8 * grp;
#pragma unroll
  for (int t = 0; t < 4; ++t) {
    const int col = t * 16 + lh;
    float s = 0.f, q = 0.f;
#pragma unroll
    for (int r = 0; r < 8; ++r) {
      const float v = acc[t][r];
      Y[(mb + r) * 64 + col] = v;
      s += v; q += v * v;
    }
    atomicAdd(&s_sum[col], s);
    atomicAdd(&s_ssq[col], q);
  }
  __syncthreads();
  if (tid < 64) { atomicAdd(&gsum[tid], s_sum[tid]); atomicAdd(&gssq[tid], s_ssq[tid]); }
}

// =====================================================================
// 5) BN helpers
// =====================================================================
__global__ void zero_stats(float* s) { if (threadIdx.x < 128) s[threadIdx.x] = 0.f; }

__global__ void bn_coeffs(const float* __restrict__ sum, const float* __restrict__ ssq,
                          const float* __restrict__ gamma, const float* __restrict__ beta,
                          float invM, float* __restrict__ A, float* __restrict__ Bc) {
  const int c = threadIdx.x;  // 64
  const float mean = sum[c] * invM;
  const float var = fmaxf(ssq[c] * invM - mean * mean, 0.f);
  const float a = gamma[c] * rsqrtf(var + 1e-5f);
  A[c] = a;
  Bc[c] = beta[c] - mean * a;
}

// mode 0: ReLU + emit next layer's f16 im2col (row pairing for stride-2 conv)
// mode 1: ReLU + emit flat f16 [r][c] (new_pts)
__global__ void bn_finalize(const float* __restrict__ Y, const float* __restrict__ A,
                            const float* __restrict__ Bc, _Float16* __restrict__ Xn,
                            long total, int Wl, int mode) {
  const long g = (long)blockIdx.x * 256 + threadIdx.x;
  if (g >= total) return;
  const int c = (int)(g & 63);
  const long r = g >> 6;
  float y = fmaf(A[c], Y[g], Bc[c]);
  y = fmaxf(y, 0.f);
  if (mode == 0) {
    const long bn = r / Wl;
    const int w = (int)(r % Wl);
    const long r2 = bn * (Wl >> 1) + (w >> 1);
    const int c2 = ((w & 1) << 6) | c;
    Xn[r2 * 128 + c2] = (_Float16)y;
  } else {
    Xn[g] = (_Float16)y;
  }
}

// final layer: BN (no relu yet) + residual add + ReLU -> d_out new_points
__global__ void residual_out(const float* __restrict__ Y, const float* __restrict__ A,
                             const float* __restrict__ Bc, const float* __restrict__ pts,
                             float* __restrict__ out) {
  const long g = (long)blockIdx.x * 256 + threadIdx.x;
  if (g >= M3 * 64) return;
  const int c = (int)(g & 63);
  const float y = fmaf(A[c], Y[g], Bc[c]) + pts[g];
  out[g] = fmaxf(y, 0.f);
}

__global__ void copy_f32(const float* __restrict__ src, float* __restrict__ dst, int n) {
  const int g = blockIdx.x * 256 + threadIdx.x;
  if (g < n) dst[g] = src[g];
}

// =====================================================================
// host-side pipeline
// =====================================================================
extern "C" void kernel_launch(void* const* d_in, const int* in_sizes, int n_in,
                              void* d_out, int out_size, void* d_ws, size_t ws_size,
                              hipStream_t stream) {
  if (ws_size < WS_NEED) return;
  const float* xyz = (const float*)d_in[0];
  const float* points = (const float*)d_in[1];
  // params flatten: conv1[0..2], conv2[0..2]; each (W, b, gamma, beta)
  const float *Wp[6], *Gp[6], *Bp[6];
  const int Cin[6] = {67, 64, 64, 67, 64, 64};
  for (int l = 0; l < 6; ++l) {
    Wp[l] = (const float*)d_in[2 + 4 * l];
    Gp[l] = (const float*)d_in[4 + 4 * l];
    Bp[l] = (const float*)d_in[5 + 4 * l];
  }

  char* ws = (char*)d_ws;
  int* idxp = (int*)(ws + OFF_IDX);
  _Float16* wts = (_Float16*)(ws + OFF_W);
  float* sum = (float*)(ws + OFF_ST);
  float* ssq = sum + 64;
  float* Ac = sum + 128;
  float* Bc = sum + 192;
  _Float16* XA = (_Float16*)(ws + OFF_XA);
  _Float16* XB = (_Float16*)(ws + OFF_XB);
  _Float16* XC = (_Float16*)(ws + OFF_XC);
  _Float16* NP = (_Float16*)(ws + OFF_NP);
  float* Y = (float*)(ws + OFF_Y);
  float* out_np = (float*)d_out + (size_t)BZ * NN * 3;

  // neighbor selection + weight prep
  select_kernel<<<dim3(NN / 256, BZ), 256, 0, stream>>>(xyz, idxp);
  for (int l = 0; l < 6; ++l)
    prep_weights<<<40, 256, 0, stream>>>(Wp[l], wts + (size_t)l * 64 * 160, Cin[l]);

  const long M[6]  = {M1, M2, M3, M1, M2, M3};
  const int  LDX[6] = {160, 128, 128, 160, 128, 128};
  _Float16* XIN[6] = {XA, XB, XC, XA, XB, XC};

  // stack 1 im2col
  im2col_first<<<(int)((M1 * 160 + 255) / 256), 256, 0, stream>>>(xyz, points, idxp, XA);

  for (int l = 0; l < 6; ++l) {
    if (l == 3)  // stack 2 im2col (gathers new_pts)
      im2col_second<<<(int)((M1 * 160 + 255) / 256), 256, 0, stream>>>(xyz, NP, idxp, XA);

    zero_stats<<<1, 128, 0, stream>>>(sum);
    gemm_bn<<<(int)(M[l] / 128), 256, 0, stream>>>(XIN[l], wts + (size_t)l * 64 * 160,
                                                   Y, sum, ssq, LDX[l], LDX[l]);
    bn_coeffs<<<1, 64, 0, stream>>>(sum, ssq, Gp[l], Bp[l], 1.0f / (float)M[l], Ac, Bc);

    const long tot = M[l] * 64;
    const int blks = (int)((tot + 255) / 256);
    if (l == 0)      bn_finalize<<<blks, 256, 0, stream>>>(Y, Ac, Bc, XB, tot, 4, 0);
    else if (l == 1) bn_finalize<<<blks, 256, 0, stream>>>(Y, Ac, Bc, XC, tot, 2, 0);
    else if (l == 2) bn_finalize<<<blks, 256, 0, stream>>>(Y, Ac, Bc, NP, tot, 1, 1);
    else if (l == 3) bn_finalize<<<blks, 256, 0, stream>>>(Y, Ac, Bc, XB, tot, 4, 0);
    else if (l == 4) bn_finalize<<<blks, 256, 0, stream>>>(Y, Ac, Bc, XC, tot, 2, 0);
    else             residual_out<<<blks, 256, 0, stream>>>(Y, Ac, Bc, points, out_np);
  }

  copy_f32<<<(BZ * NN * 3 + 255) / 256, 256, 0, stream>>>(xyz, (float*)d_out, BZ * NN * 3);
}
